// ConditionedPNA_6012954214821
// MI455X (gfx1250) — compile-verified, matrix operand order checked by reading, not run on hardware
//
#include <hip/hip_runtime.h>
#include <hip/hip_bf16.h>

// ---------------- problem constants (match reference) ----------------
#define Bq      2
#define NEGQ    16
#define Nn      10000
#define Ne      80000
#define Dd      128
#define NUMREL  237
#define Lp      3
#define Kk      1000           // top-k nodes per graph
#define BN      (Bq*Nn)        // 20000
#define BE      (Bq*Ne)        // 160000
#define EPSV    1e-5f

typedef __bf16 bf16;
typedef __attribute__((ext_vector_type(16))) __bf16 v16bf;
typedef __attribute__((ext_vector_type(8)))  float  v8f;

#define WMMA_BF16(a,b,c) __builtin_amdgcn_wmma_f32_16x16x32_bf16(false,(a),false,(b),(short)0,(c),false,false)

// order-preserving float->uint key (larger float -> larger key)
__device__ __forceinline__ unsigned f2key(float f){
  unsigned u = __float_as_uint(f);
  return (u & 0x80000000u) ? ~u : (u | 0x80000000u);
}
__device__ __forceinline__ void atomicMaxF(float* a, float v){
  if (v >= 0.f) atomicMax((int*)a, __float_as_int(v));
  else          atomicMin((unsigned int*)a, __float_as_uint(v));
}
__device__ __forceinline__ void atomicMinF(float* a, float v){
  if (v >= 0.f) atomicMin((int*)a, __float_as_int(v));
  else          atomicMax((unsigned int*)a, __float_as_uint(v));
}

// ---------------- weight prep: WT[n*K+k] = bf16(W[k*N+n]) ----------------
__global__ void k_transpose_bf16(const float* __restrict__ W, bf16* __restrict__ WT,
                                 int K, int Ncols){
  int i = blockIdx.x*256 + threadIdx.x;
  if (i >= K*Ncols) return;
  int k = i / Ncols, n = i - k*Ncols;
  WT[(size_t)n*K + k] = (bf16)W[i];
}

// ---------------- setup: index rewiring + quirk rows ----------------
__global__ void k_setup(const int* __restrict__ h_index, const int* __restrict__ r_index,
                        const int* __restrict__ t_index, const float* __restrict__ hidden_states,
                        const float* __restrict__ rel_embedding,
                        int* flat_h, int* flat_t, int* t_sel,
                        float* relE, float* h_vals, float* t_vals){
  __shared__ int sh_r[Bq]; __shared__ int sh_hq, sh_tq;
  if (threadIdx.x == 0){
    for (int b = 0; b < Bq; ++b){
      int h0 = h_index[b*NEGQ]; bool neg = true;
      for (int j = 1; j < NEGQ; ++j) if (h_index[b*NEGQ+j] != h0) neg = false;
      int hh = neg ? h_index[b*NEGQ] : t_index[b*NEGQ];
      flat_h[b] = hh + b*Nn;
      for (int j = 0; j < NEGQ; ++j)
        t_sel[b*NEGQ+j] = neg ? t_index[b*NEGQ+j] : h_index[b*NEGQ+j];
      flat_t[b] = t_sel[b*NEGQ] + b*Nn;
      sh_r[b] = r_index[b*NEGQ] + (neg ? 0 : NUMREL);
      if (b == 0){ sh_hq = hh; sh_tq = t_sel[0]; }
    }
  }
  __syncthreads();
  int d = threadIdx.x;
  if (d < Dd){
    // literal repeat_interleave quirk: row 0 of gathered embeddings for ALL batches
    h_vals[d] = hidden_states[(size_t)sh_hq*Dd + d];
    t_vals[d] = hidden_states[(size_t)sh_tq*Dd + d];
    for (int b = 0; b < Bq; ++b)
      relE[b*Dd+d] = rel_embedding[(size_t)sh_r[b]*Dd + d];
  }
}

// rc[b][col] = rel_b @ lin_W[128:256] + lin_b   (constant per graph, all layers)
__global__ void k_rc(const float* __restrict__ lin_W, const float* __restrict__ lin_b,
                     const float* __restrict__ relE, float* rc){
  int b = blockIdx.x, col = threadIdx.x;
  float acc = lin_b[col];
  for (int k = 0; k < Dd; ++k)
    acc += relE[b*Dd+k] * lin_W[(size_t)(Dd+k)*Dd + col];
  rc[b*Dd+col] = acc;
}

__global__ void k_init_hidden(float* hidden, const int* flat_h, const int* flat_t,
                              const float* h_vals, const float* t_vals){
  int i = blockIdx.x*256 + threadIdx.x;
  if (i >= BN*Dd) return;
  int node = i >> 7, col = i & 127;
  float v = 0.f;
  if (node == flat_t[0] || node == flat_t[1]) v = t_vals[col];   // .at[flat_t].set
  if (node == flat_h[0] || node == flat_h[1]) v = h_vals[col];   // then .at[flat_h].set
  hidden[i] = v;
}

__global__ void k_zero_score(float* score){
  int i = blockIdx.x*256 + threadIdx.x;
  if (i < BN) score[i] = 0.f;
}

// scalar _score for the 2 head rows (fp32 exact path, tiny)
__global__ void k_head_score(const float* __restrict__ lin_W, const float* __restrict__ m1_W,
                             const float* __restrict__ m1_b, const float* __restrict__ m2_W,
                             const float* __restrict__ m2_b, const float* __restrict__ rc,
                             const float* __restrict__ h_vals, const int* __restrict__ flat_h,
                             float* score){
  __shared__ float sx[Dd], sh2[Dd];
  int b = blockIdx.x, col = threadIdx.x;
  float heur = rc[b*Dd+col];
  for (int k = 0; k < Dd; ++k) heur += h_vals[k] * lin_W[(size_t)k*Dd + col];
  sx[col] = heur * h_vals[col];
  __syncthreads();
  float acc = m1_b[col];
  for (int k = 0; k < Dd; ++k) acc += sx[k] * m1_W[(size_t)k*Dd + col];
  sh2[col] = fmaxf(acc, 0.f);
  __syncthreads();
  if (col == 0){
    float s = m2_b[0];
    for (int k = 0; k < Dd; ++k) s += sh2[k] * m2_W[k];
    score[flat_h[b]] = s;
  }
}

// ---------------- exact top-K threshold: MSB-first radix select ----------------
__global__ void k_radix_select(const float* __restrict__ score, unsigned* Tkey, int* Gcnt){
  int b = blockIdx.x;
  const float* s = score + b*Nn;
  __shared__ int red[256];
  unsigned prefix = 0; int remaining = Kk;
  for (int bit = 31; bit >= 0; --bit){
    unsigned target = (prefix >> bit) | 1u;
    int c = 0;
    for (int i = threadIdx.x; i < Nn; i += 256)
      if ((f2key(s[i]) >> bit) == target) c++;
    red[threadIdx.x] = c; __syncthreads();
    for (int st = 128; st > 0; st >>= 1){
      if (threadIdx.x < st) red[threadIdx.x] += red[threadIdx.x+st];
      __syncthreads();
    }
    int tot = red[0]; __syncthreads();
    if (tot >= remaining) prefix |= (1u << bit);
    else                  remaining -= tot;
  }
  int c = 0;
  for (int i = threadIdx.x; i < Nn; i += 256) if (f2key(s[i]) > prefix) c++;
  red[threadIdx.x] = c; __syncthreads();
  for (int st = 128; st > 0; st >>= 1){
    if (threadIdx.x < st) red[threadIdx.x] += red[threadIdx.x+st];
    __syncthreads();
  }
  if (threadIdx.x == 0){ Tkey[b] = prefix; Gcnt[b] = red[0]; }
}

__global__ void k_mark(const float* __restrict__ score, const unsigned* __restrict__ Tkey,
                       int* node_sel){
  int i = blockIdx.x*256 + threadIdx.x;
  if (i >= BN) return;
  int g = i / Nn;
  node_sel[i] = (f2key(score[i]) > Tkey[g]) ? 1 : 0;
}

// jax top_k tie-break: lowest index first among values == threshold
__global__ void k_tiefill(const float* __restrict__ score, const unsigned* __restrict__ Tkey,
                          const int* __restrict__ Gcnt, int* node_sel){
  if (threadIdx.x != 0) return;
  int b = blockIdx.x; unsigned T = Tkey[b]; int total = Gcnt[b];
  for (int i = 0; i < Nn && total < Kk; ++i){
    if (f2key(score[b*Nn+i]) == T){ node_sel[b*Nn+i] = 1; total++; }
  }
}

__global__ void k_zero_agg(float* s1, float* s2, float* mx, float* mn,
                           float* cnt_dst, float* out_deg, int* ec, float* logsum){
  int i = blockIdx.x*256 + threadIdx.x;
  if (i < BN*Dd){ s1[i]=0.f; s2[i]=0.f; mx[i]=-INFINITY; mn[i]=INFINITY; }
  if (i < BN){ cnt_dst[i]=0.f; out_deg[i]=0.f; }
  if (i == 0){ *ec = 0; *logsum = 0.f; }
}

__global__ void k_edge_mask(const int* __restrict__ edge_index, const int* __restrict__ node_sel,
                            int* sel_edges, int* ec, float* out_deg, float* cnt_dst){
  int e = blockIdx.x*256 + threadIdx.x;
  if (e >= BE) return;
  int g = e / Ne, le = e - g*Ne;
  int s = edge_index[le] + g*Nn;
  if (!node_sel[s]) return;
  int d = edge_index[Ne + le] + g*Nn;
  int pos = atomicAdd(ec, 1);
  sel_edges[pos] = e;
  atomicAdd(&out_deg[s], 1.f);
  atomicAdd(&cnt_dst[d], 1.f);
}

// ---------------- edge GEMM (WMMA bf16) fused with scatter-aggregation ----------------
// m = [h_dst | h_src](16x256) @ pre_W(256x128) + pre_b ; scatter s1+=m, s2+=m^2, max, min
// A tile staged global->LDS with async b128 transfers (ASYNCcnt path).
__global__ __launch_bounds__(128)
void k_edge_gemm(const bf16* __restrict__ hbf, const int* __restrict__ edge_index,
                 const int* __restrict__ sel_edges, const int* __restrict__ ec,
                 const bf16* __restrict__ preWT, const float* __restrict__ pre_b,
                 float* s1, float* s2, float* mx, float* mn){
  __shared__ bf16 Xs[16][256];           // 16 edges x [h_dst(128) | h_src(128)]
  __shared__ int dstN[16], srcN[16];
  int base = blockIdx.x * 16;
  int M = *ec;
  if (base >= M) return;
  if (threadIdx.x < 16){
    int r = threadIdx.x, dd = -1, ss = -1;
    if (base + r < M){
      int e = sel_edges[base + r];
      int g = e / Ne, le = e - g*Ne;
      ss = edge_index[le]      + g*Nn;
      dd = edge_index[Ne + le] + g*Nn;
    }
    dstN[r] = dd; srcN[r] = ss;
  }
  __syncthreads();
  // zero-fill padding rows with plain LDS stores (disjoint from async targets)
  for (int i = threadIdx.x; i < 16*256; i += 128){
    int r = i >> 8;
    if (dstN[r] < 0) Xs[r][i & 255] = (bf16)0.f;
  }
  // async global->LDS: 512 x 16B chunks, 4 per thread
  unsigned ldsBase = (unsigned)(unsigned long long)(&Xs[0][0]); // LDS offset = addr[31:0]
  #pragma unroll
  for (int c4 = 0; c4 < 4; ++c4){
    int chunk  = c4*128 + threadIdx.x;
    int r      = chunk >> 5;          // row (512 B each)
    int within = chunk & 31;          // 16B chunk within row
    int node   = (within & 16) ? srcN[r] : dstN[r];
    if (node >= 0){
      unsigned ldsAddr = ldsBase + (unsigned)(r*512 + within*16);
      const bf16* gp = hbf + (size_t)node*Dd + (size_t)(within & 15)*8;
      asm volatile("global_load_async_to_lds_b128 %0, %1, off"
                   :: "v"(ldsAddr), "v"((unsigned long long)gp) : "memory");
    }
  }
  asm volatile("s_wait_asynccnt 0x0" ::: "memory");
  __syncthreads();

  int lane = threadIdx.x & 31, wave = threadIdx.x >> 5;
  int m = lane & 15, hi = lane >> 4;
  int col0 = wave*32 + m, col1 = col0 + 16;
  v8f acc0 = {0.f,0.f,0.f,0.f,0.f,0.f,0.f,0.f};
  v8f acc1 = acc0;
  #pragma unroll
  for (int ks = 0; ks < 8; ++ks){
    int k0 = ks*32 + hi*8;
    v16bf a, b0, b1;
    #pragma unroll
    for (int i = 0; i < 8; ++i){ a[i] = Xs[m][k0+i]; a[8+i] = Xs[m][k0+16+i]; }
    const bf16* w0 = preWT + (size_t)col0*256 + k0;
    const bf16* w1 = preWT + (size_t)col1*256 + k0;
    #pragma unroll
    for (int i = 0; i < 8; ++i){
      b0[i] = w0[i]; b0[8+i] = w0[16+i];
      b1[i] = w1[i]; b1[8+i] = w1[16+i];
    }
    acc0 = WMMA_BF16(a, b0, acc0);   // two independent XDL chains
    acc1 = WMMA_BF16(a, b1, acc1);
  }
  float pb0 = pre_b[col0], pb1 = pre_b[col1];
  #pragma unroll
  for (int r = 0; r < 8; ++r){
    int row = r + hi*8;
    int dd = dstN[row];
    if (dd >= 0){
      float v0 = acc0[r] + pb0;
      float v1 = acc1[r] + pb1;
      size_t o0 = (size_t)dd*Dd + col0;
      size_t o1 = (size_t)dd*Dd + col1;
      atomicAdd(&s1[o0], v0);  atomicAdd(&s2[o0], v0*v0);
      atomicMaxF(&mx[o0], v0); atomicMinF(&mn[o0], v0);
      atomicAdd(&s1[o1], v1);  atomicAdd(&s2[o1], v1*v1);
      atomicMaxF(&mx[o1], v1); atomicMinF(&mn[o1], v1);
    }
  }
}

// s1 -> mean, s2 -> std, zero max/min for empty segments, log-degree + reduction
__global__ void k_finalize(float* s1, float* s2, float* mx, float* mn,
                           const float* __restrict__ cnt_dst, const float* __restrict__ out_deg,
                           float* logd, float* logsum){
  int i = blockIdx.x*256 + threadIdx.x;
  if (i >= BN*Dd) return;
  int node = i >> 7, col = i & 127;
  float c = cnt_dst[node];
  float denom = fmaxf(c, 1.f);
  float mean = s1[i] / denom;
  float var  = fmaxf(s2[i] / denom - mean*mean, 0.f);
  s1[i] = mean;
  s2[i] = sqrtf(var + EPSV);
  if (c <= 0.f){ mx[i] = 0.f; mn[i] = 0.f; }
  if (col == 0){
    float dg = fmaxf(out_deg[node], 1.f);
    float ld = logf(dg + 1.f);
    logd[node] = ld;
    atomicAdd(logsum, ld);
  }
}

__global__ void k_avg(const float* logsum, float* avg){
  if (blockIdx.x == 0 && threadIdx.x == 0) *avg = *logsum / (float)BN;
}

// ---------------- post GEMM: K=512 vs 3 slices of post_W, scaler epilogue ----------------
__global__ __launch_bounds__(128)
void k_post_gemm(const float* __restrict__ meanA, const float* __restrict__ mxA,
                 const float* __restrict__ mnA,   const float* __restrict__ stdA,
                 const bf16* __restrict__ postWT, const float* __restrict__ post_b,
                 const float* __restrict__ logd,  const float* __restrict__ avgp,
                 float* Cout){
  __shared__ bf16 As[16][512];
  int node0 = blockIdx.x * 16;
  for (int i = threadIdx.x; i < 16*512; i += 128){
    int r = i >> 9, k = i & 511;
    int seg = k >> 7, c = k & 127;
    const float* src = (seg==0) ? meanA : (seg==1) ? mxA : (seg==2) ? mnA : stdA;
    As[r][k] = (bf16)src[(size_t)(node0+r)*Dd + c];
  }
  __syncthreads();
  float avg = *avgp;
  int lane = threadIdx.x & 31, wave = threadIdx.x >> 5;
  int m = lane & 15, hi = lane >> 4;
  int col0 = wave*32 + m, col1 = col0 + 16;
  v8f p00 = {0.f,0.f,0.f,0.f,0.f,0.f,0.f,0.f};
  v8f p01 = p00, p02 = p00, p10 = p00, p11 = p00, p12 = p00;
  for (int ks = 0; ks < 16; ++ks){
    int k0 = ks*32 + hi*8;
    v16bf a;
    #pragma unroll
    for (int i = 0; i < 8; ++i){ a[i] = As[m][k0+i]; a[8+i] = As[m][k0+16+i]; }
    const bf16* w0 = postWT + (size_t)col0*1536 + k0;
    const bf16* w1 = postWT + (size_t)col1*1536 + k0;
    v16bf b00, b01, b02, b10, b11, b12;
    #pragma unroll
    for (int i = 0; i < 8; ++i){
      b00[i] = w0[i];       b00[8+i] = w0[16+i];
      b01[i] = w0[512+i];   b01[8+i] = w0[512+16+i];
      b02[i] = w0[1024+i];  b02[8+i] = w0[1024+16+i];
      b10[i] = w1[i];       b10[8+i] = w1[16+i];
      b11[i] = w1[512+i];   b11[8+i] = w1[512+16+i];
      b12[i] = w1[1024+i];  b12[8+i] = w1[1024+16+i];
    }
    p00 = WMMA_BF16(a, b00, p00);   // 6 independent accumulator chains
    p01 = WMMA_BF16(a, b01, p01);
    p02 = WMMA_BF16(a, b02, p02);
    p10 = WMMA_BF16(a, b10, p10);
    p11 = WMMA_BF16(a, b11, p11);
    p12 = WMMA_BF16(a, b12, p12);
  }
  float pb0 = post_b[col0], pb1 = post_b[col1];
  #pragma unroll
  for (int r = 0; r < 8; ++r){
    int node = node0 + r + hi*8;
    float ld = logd[node];
    float asc = ld / avg, bsc = avg / ld;
    Cout[(size_t)node*Dd + col0] = p00[r] + asc*p01[r] + bsc*p02[r] + pb0;
    Cout[(size_t)node*Dd + col1] = p10[r] + asc*p11[r] + bsc*p12[r] + pb1;
  }
}

// ---------------- generic node GEMM (K=128) with epilogue modes ----------------
// mode 0: C = acc + bias ; mode 1: C = (acc + rc[graph]) * hidden ; mode 2: C = relu(acc + bias)
__global__ __launch_bounds__(128)
void k_node_gemm(const float* __restrict__ A, const bf16* __restrict__ WT,
                 const float* __restrict__ bias, float* C, int mode,
                 const float* __restrict__ hiddenF, const float* __restrict__ rc){
  __shared__ bf16 As[16][128];
  int node0 = blockIdx.x * 16;
  for (int i = threadIdx.x; i < 16*128; i += 128){
    int r = i >> 7, c = i & 127;
    As[r][c] = (bf16)A[(size_t)(node0+r)*Dd + c];
  }
  __syncthreads();
  int lane = threadIdx.x & 31, wave = threadIdx.x >> 5;
  int m = lane & 15, hi = lane >> 4;
  int col0 = wave*32 + m, col1 = col0 + 16;
  v8f acc0 = {0.f,0.f,0.f,0.f,0.f,0.f,0.f,0.f};
  v8f acc1 = acc0;
  #pragma unroll
  for (int ks = 0; ks < 4; ++ks){
    int k0 = ks*32 + hi*8;
    v16bf a, b0, b1;
    #pragma unroll
    for (int i = 0; i < 8; ++i){ a[i] = As[m][k0+i]; a[8+i] = As[m][k0+16+i]; }
    const bf16* w0 = WT + (size_t)col0*Dd + k0;
    const bf16* w1 = WT + (size_t)col1*Dd + k0;
    #pragma unroll
    for (int i = 0; i < 8; ++i){
      b0[i] = w0[i]; b0[8+i] = w0[16+i];
      b1[i] = w1[i]; b1[8+i] = w1[16+i];
    }
    acc0 = WMMA_BF16(a, b0, acc0);
    acc1 = WMMA_BF16(a, b1, acc1);
  }
  #pragma unroll
  for (int r = 0; r < 8; ++r){
    int node = node0 + r + hi*8;
    float v0 = acc0[r], v1 = acc1[r];
    if (mode == 0){ v0 += bias[col0]; v1 += bias[col1]; }
    else if (mode == 1){
      int g = node / Nn;
      v0 = (v0 + rc[g*Dd+col0]) * hiddenF[(size_t)node*Dd+col0];
      v1 = (v1 + rc[g*Dd+col1]) * hiddenF[(size_t)node*Dd+col1];
    } else {
      v0 = fmaxf(v0 + bias[col0], 0.f);
      v1 = fmaxf(v1 + bias[col1], 0.f);
    }
    C[(size_t)node*Dd + col0] = v0;
    C[(size_t)node*Dd + col1] = v1;
  }
}

__global__ void k_residual(float* hidden, const float* __restrict__ addv,
                           const float* __restrict__ out_deg){
  int i = blockIdx.x*256 + threadIdx.x;
  if (i >= BN*Dd) return;
  if (out_deg[i >> 7] > 0.f) hidden[i] += addv[i];
}

__global__ void k_tobf(const float* __restrict__ hidden, bf16* hbf){
  int i = blockIdx.x*256 + threadIdx.x;
  if (i < BN*Dd) hbf[i] = (bf16)hidden[i];
}

__global__ void k_score_gemv(const float* __restrict__ H2, const float* __restrict__ m2_W,
                             const float* __restrict__ m2_b, float* score){
  int node = blockIdx.x*256 + threadIdx.x;
  if (node >= BN) return;
  float acc = m2_b[0];
  const float* row = H2 + (size_t)node*Dd;
  for (int k = 0; k < Dd; ++k) acc += row[k] * m2_W[k];
  score[node] = acc;
}

__global__ void k_gather_out(const float* __restrict__ score, const int* __restrict__ t_sel,
                             float* outp){
  int t = threadIdx.x;
  if (t < Bq*NEGQ){
    int b = t / NEGQ;
    outp[t] = score[b*Nn + t_sel[t]];
  }
}

// ---------------- host orchestration ----------------
extern "C" void kernel_launch(void* const* d_in, const int* in_sizes, int n_in,
                              void* d_out, int out_size, void* d_ws, size_t ws_size,
                              hipStream_t stream){
  (void)in_sizes; (void)n_in; (void)out_size; (void)ws_size;
  const int*   h_index       = (const int*)  d_in[0];
  const int*   r_index       = (const int*)  d_in[1];
  const int*   t_index       = (const int*)  d_in[2];
  const float* hidden_states = (const float*)d_in[3];
  const int*   edge_index    = (const int*)  d_in[4];
  const float* rel_embedding = (const float*)d_in[5];
  const float* lin_W  = (const float*)d_in[6];
  const float* lin_b  = (const float*)d_in[7];
  const float* m1_W   = (const float*)d_in[8];
  const float* m1_b   = (const float*)d_in[9];
  const float* m2_W   = (const float*)d_in[10];
  const float* m2_b   = (const float*)d_in[11];
  const float* pre_W  = (const float*)d_in[12];
  const float* pre_b  = (const float*)d_in[13];
  const float* post_W = (const float*)d_in[14];
  const float* post_b = (const float*)d_in[15];
  const float* out_W  = (const float*)d_in[16];
  const float* out_b  = (const float*)d_in[17];
  float* outp = (float*)d_out;

  char* wptr = (char*)d_ws;
  auto alloc = [&](size_t bytes) -> void* {
    void* p = (void*)wptr;
    wptr += (bytes + 255) & ~(size_t)255;
    return p;
  };
  float* hidden  = (float*)alloc((size_t)BN*Dd*4);
  float* tmpA    = (float*)alloc((size_t)BN*Dd*4);
  float* tmpB    = (float*)alloc((size_t)BN*Dd*4);
  float* s1      = (float*)alloc((size_t)BN*Dd*4);
  float* s2      = (float*)alloc((size_t)BN*Dd*4);
  float* mx      = (float*)alloc((size_t)BN*Dd*4);
  float* mn      = (float*)alloc((size_t)BN*Dd*4);
  bf16*  hbf     = (bf16*) alloc((size_t)BN*Dd*2);
  bf16*  preWT   = (bf16*) alloc((size_t)Lp*Dd*256*2);
  bf16*  postWT  = (bf16*) alloc((size_t)Lp*Dd*1536*2);
  bf16*  outWT   = (bf16*) alloc((size_t)Lp*Dd*Dd*2);
  bf16*  linW0T  = (bf16*) alloc((size_t)Dd*Dd*2);
  bf16*  m1WT    = (bf16*) alloc((size_t)Dd*Dd*2);
  float* score   = (float*)alloc((size_t)BN*4);
  float* cnt_dst = (float*)alloc((size_t)BN*4);
  float* out_deg = (float*)alloc((size_t)BN*4);
  float* logd    = (float*)alloc((size_t)BN*4);
  int*   node_sel= (int*)  alloc((size_t)BN*4);
  int*   sel_edges=(int*)  alloc((size_t)BE*4);
  float* rc      = (float*)alloc((size_t)Bq*Dd*4);
  float* relE    = (float*)alloc((size_t)Bq*Dd*4);
  float* h_vals  = (float*)alloc((size_t)Dd*4);
  float* t_vals  = (float*)alloc((size_t)Dd*4);
  int*   flat_h  = (int*)  alloc(64);
  int*   flat_t  = (int*)  alloc(64);
  int*   t_sel   = (int*)  alloc((size_t)Bq*NEGQ*4);
  unsigned* Tkey = (unsigned*)alloc(64);
  int*   Gcnt    = (int*)  alloc(64);
  int*   ec      = (int*)  alloc(64);
  float* logsum  = (float*)alloc(64);
  float* avg     = (float*)alloc(64);

  // weight prep (bf16 transposed so B fragments are contiguous 16B loads)
  for (int l = 0; l < Lp; ++l){
    k_transpose_bf16<<<(256*Dd+255)/256,256,0,stream>>>(pre_W  + (size_t)l*256*Dd,  preWT  + (size_t)l*Dd*256,  256,  Dd);
    k_transpose_bf16<<<(1536*Dd+255)/256,256,0,stream>>>(post_W + (size_t)l*1536*Dd, postWT + (size_t)l*Dd*1536, 1536, Dd);
    k_transpose_bf16<<<(Dd*Dd+255)/256,256,0,stream>>>(out_W  + (size_t)l*Dd*Dd,   outWT  + (size_t)l*Dd*Dd,   Dd,   Dd);
  }
  k_transpose_bf16<<<(Dd*Dd+255)/256,256,0,stream>>>(lin_W, linW0T, Dd, Dd);  // rows 0..127 only
  k_transpose_bf16<<<(Dd*Dd+255)/256,256,0,stream>>>(m1_W,  m1WT,   Dd, Dd);

  k_setup<<<1,128,0,stream>>>(h_index, r_index, t_index, hidden_states, rel_embedding,
                              flat_h, flat_t, t_sel, relE, h_vals, t_vals);
  k_rc<<<Bq,128,0,stream>>>(lin_W, lin_b, relE, rc);

  int gBND = (BN*Dd + 255)/256;
  k_init_hidden<<<gBND,256,0,stream>>>(hidden, flat_h, flat_t, h_vals, t_vals);
  k_tobf<<<gBND,256,0,stream>>>(hidden, hbf);
  k_zero_score<<<(BN+255)/256,256,0,stream>>>(score);
  k_head_score<<<Bq,128,0,stream>>>(lin_W, m1_W, m1_b, m2_W, m2_b, rc, h_vals, flat_h, score);

  for (int l = 0; l < Lp; ++l){
    // select_edges: per-graph exact top-K
    k_radix_select<<<Bq,256,0,stream>>>(score, Tkey, Gcnt);
    k_mark<<<(BN+255)/256,256,0,stream>>>(score, Tkey, node_sel);
    k_tiefill<<<Bq,32,0,stream>>>(score, Tkey, Gcnt, node_sel);
    // aggregation buffers + edge compaction + degrees
    k_zero_agg<<<gBND,256,0,stream>>>(s1, s2, mx, mn, cnt_dst, out_deg, ec, logsum);
    k_edge_mask<<<(BE+255)/256,256,0,stream>>>(edge_index, node_sel, sel_edges, ec, out_deg, cnt_dst);
    // edge pre-linear (WMMA) fused with scatter aggregation
    k_edge_gemm<<<(BE+15)/16,128,0,stream>>>(hbf, edge_index, sel_edges, ec,
                                             preWT + (size_t)l*Dd*256, pre_b + l*Dd,
                                             s1, s2, mx, mn);
    k_finalize<<<gBND,256,0,stream>>>(s1, s2, mx, mn, cnt_dst, out_deg, logd, logsum);
    k_avg<<<1,1,0,stream>>>(logsum, avg);
    // PNA post: 3x(512-K) WMMA accumulators + scaler epilogue, then out GEMM
    k_post_gemm<<<BN/16,128,0,stream>>>(s1, mx, mn, s2, postWT + (size_t)l*Dd*1536,
                                        post_b + l*Dd, logd, avg, tmpA);
    k_node_gemm<<<BN/16,128,0,stream>>>(tmpA, outWT + (size_t)l*Dd*Dd, out_b + l*Dd,
                                        tmpB, 0, nullptr, nullptr);
    k_residual<<<gBND,256,0,stream>>>(hidden, tmpB, out_deg);
    k_tobf<<<gBND,256,0,stream>>>(hidden, hbf);
    // score MLP over all nodes
    k_node_gemm<<<BN/16,128,0,stream>>>(hidden, linW0T, nullptr, tmpA, 1, hidden, rc);
    k_node_gemm<<<BN/16,128,0,stream>>>(tmpA, m1WT, m1_b, tmpB, 2, nullptr, nullptr);
    k_score_gemv<<<(BN+255)/256,256,0,stream>>>(tmpB, m2_W, m2_b, score);
  }
  k_gather_out<<<1,64,0,stream>>>(score, t_sel, outp);
}
